// AttentionBlock_35347580846818
// MI455X (gfx1250) — matrix operand order repaired and checked by reading, not compiled, not run on hardware
//
#include <hip/hip_runtime.h>
#include <math.h>

#define BATCH 4
#define NPIX  4096          // H*W
#define CH    512
#define NG    32
#define CPG   16            // CH/NG
#define GN_EPS 1e-5f
#define QSCALE 0.044194173824159216f   // 1/sqrt(512)

typedef __bf16 bf16;
typedef __attribute__((ext_vector_type(16))) __bf16 v16bf;
typedef __attribute__((ext_vector_type(8)))  __bf16 v8bf;
typedef __attribute__((ext_vector_type(8)))  float  v8f;
typedef __attribute__((ext_vector_type(4)))  int    v4i;

// Async global->LDS copy (gfx1250 GLOBAL_LOAD_ASYNC_TO_LDS_B128), guarded so
// the file still compiles if the toolchain lacks the builtin.
// Probe-learned signature: (v4i addrspace(1)*, v4i addrspace(3)*, imm, imm).
#if defined(__has_builtin)
#if __has_builtin(__builtin_amdgcn_global_load_async_to_lds_b128) && \
    __has_builtin(__builtin_amdgcn_s_wait_asynccnt)
#define HAVE_ASYNC_LDS 1
#endif
#endif
#ifndef HAVE_ASYNC_LDS
#define HAVE_ASYNC_LDS 0
#endif

#if HAVE_ASYNC_LDS
typedef __attribute__((address_space(1))) v4i* as1_v4i_p;
typedef __attribute__((address_space(3))) v4i* as3_v4i_p;
#endif

static __device__ __forceinline__ bf16 f2bf(float f) {
  union { float f; unsigned u; } a; a.f = f;
  unsigned r = a.u + 0x7FFFu + ((a.u >> 16) & 1u);   // round-to-nearest-even
  unsigned short h = (unsigned short)(r >> 16);
  bf16 o; __builtin_memcpy(&o, &h, 2); return o;
}

// ---- WMMA fragment loaders (wave32 layouts per CDNA5 ISA 7.12.2) ----
// All loaders use two 16-byte vector loads per lane (no element packing).

// A fragment: 16(M) x 32(K) bf16 from row-major source, leading dim ld (elements).
// lanes 0-15: row M=row0+lane, K={k0..k0+7, k0+16..k0+23}; lanes 16-31: K +8.
template <typename PT>
static __device__ __forceinline__ v16bf load_a_frag(PT p, int ld, int row0, int k0) {
  const int lane = threadIdx.x & 31;
  const int hl   = lane >> 4;
  const int r    = lane & 15;
  PT q = p + (size_t)(row0 + r) * ld + k0 + hl * 8;
  const v8bf lo = *(const v8bf*)q;
  const v8bf hi = *(const v8bf*)(q + 16);
  return __builtin_shufflevector(lo, hi, 0, 1, 2, 3, 4, 5, 6, 7,
                                 8, 9, 10, 11, 12, 13, 14, 15);
}

// B fragment: 32(K) x 16(N) bf16 from an "n-major" source: element (k, n) is
// stored at p[n*ld + k].  Lane n reads K={k0..k0+7, k0+16..k0+23} (+8 for lanes 16-31).
template <typename PT>
static __device__ __forceinline__ v16bf load_b_frag_nmaj(PT p, int ld, int k0, int n0) {
  const int lane = threadIdx.x & 31;
  const int hl   = lane >> 4;
  const int n    = lane & 15;
  PT q = p + (size_t)(n0 + n) * ld + k0 + 8 * hl;
  const v8bf lo = *(const v8bf*)q;
  const v8bf hi = *(const v8bf*)(q + 16);
  return __builtin_shufflevector(lo, hi, 0, 1, 2, 3, 4, 5, 6, 7,
                                 8, 9, 10, 11, 12, 13, 14, 15);
}

static __device__ __forceinline__ v8f wmma_bf16(v16bf a, v16bf b, v8f c) {
  return __builtin_amdgcn_wmma_f32_16x16x32_bf16(false, a, false, b, (short)0, c, false, false);
}

// ---------------------------------------------------------------------------
// Kernel 1: GroupNorm statistics. One block per (batch, group).
// ---------------------------------------------------------------------------
__global__ void k_gn_stats(const float* __restrict__ x, float* __restrict__ stats) {
  const int bg = blockIdx.x;
  const int b  = bg >> 5;
  const int g  = bg & 31;
  const float* xb = x + (size_t)b * NPIX * CH + g * CPG;

  float s = 0.f, ss = 0.f;
  for (int t = threadIdx.x; t < NPIX * 4; t += blockDim.x) {
    const int n = t >> 2, c4 = t & 3;
    const float4 v = *(const float4*)(xb + (size_t)n * CH + c4 * 4);
    s  += v.x + v.y + v.z + v.w;
    ss += v.x * v.x + v.y * v.y + v.z * v.z + v.w * v.w;
  }
  __shared__ float rs[256], rss[256];
  rs[threadIdx.x] = s; rss[threadIdx.x] = ss;
  __syncthreads();
  for (int off = 128; off > 0; off >>= 1) {
    if ((int)threadIdx.x < off) {
      rs[threadIdx.x]  += rs[threadIdx.x + off];
      rss[threadIdx.x] += rss[threadIdx.x + off];
    }
    __syncthreads();
  }
  if (threadIdx.x == 0) {
    const float inv = 1.0f / (float)(NPIX * CPG);
    const float m = rs[0] * inv;
    const float var = rss[0] * inv - m * m;
    stats[bg * 2 + 0] = m;
    stats[bg * 2 + 1] = rsqrtf(var + GN_EPS);
  }
}

// ---------------------------------------------------------------------------
// Kernel 2: convert the four 512x512 weight matrices to bf16, TRANSPOSED:
// wT[d*CH + c] = W[c*CH + d], so B-fragments are n-major contiguous.
// ---------------------------------------------------------------------------
__global__ void k_convert_w(const float* __restrict__ wq, const float* __restrict__ wk,
                            const float* __restrict__ wv, const float* __restrict__ wp,
                            bf16* __restrict__ oq, bf16* __restrict__ ok,
                            bf16* __restrict__ ov, bf16* __restrict__ opp) {
  const float* src; bf16* dst;
  switch (blockIdx.y) {
    case 0:  src = wq; dst = oq;  break;
    case 1:  src = wk; dst = ok;  break;
    case 2:  src = wv; dst = ov;  break;
    default: src = wp; dst = opp; break;
  }
  const int i = blockIdx.x * blockDim.x + threadIdx.x;   // output index
  if (i < CH * CH) {
    const int d = i >> 9, c = i & (CH - 1);
    dst[i] = f2bf(src[(size_t)c * CH + d]);
  }
}

// ---------------------------------------------------------------------------
// Kernel 3: apply GroupNorm, emit bf16 activations xnb[b][n][c].
// One thread per 8-channel segment (segments never straddle a group).
// grid = BATCH*NPIX*64 / 256
// ---------------------------------------------------------------------------
__global__ void k_gn_apply(const float* __restrict__ x, const float* __restrict__ stats,
                           const float* __restrict__ gamma, const float* __restrict__ beta,
                           bf16* __restrict__ xnb) {
  const unsigned i = blockIdx.x * blockDim.x + threadIdx.x;   // segment id
  const int b   = i >> 18;                 // 262144 segments per batch
  const int rem = i & 262143;
  const int n   = rem >> 6, seg = rem & 63;
  const int ch0 = seg * 8;
  const int g   = ch0 >> 4;
  const float mu = stats[(b * NG + g) * 2 + 0];
  const float rs = stats[(b * NG + g) * 2 + 1];
  const float* xp = x + ((size_t)b * NPIX + n) * CH + ch0;
  const float4 x0 = *(const float4*)xp;
  const float4 x1 = *(const float4*)(xp + 4);
  v8bf o8;
  o8[0] = f2bf((x0.x - mu) * rs * gamma[ch0 + 0] + beta[ch0 + 0]);
  o8[1] = f2bf((x0.y - mu) * rs * gamma[ch0 + 1] + beta[ch0 + 1]);
  o8[2] = f2bf((x0.z - mu) * rs * gamma[ch0 + 2] + beta[ch0 + 2]);
  o8[3] = f2bf((x0.w - mu) * rs * gamma[ch0 + 3] + beta[ch0 + 3]);
  o8[4] = f2bf((x1.x - mu) * rs * gamma[ch0 + 4] + beta[ch0 + 4]);
  o8[5] = f2bf((x1.y - mu) * rs * gamma[ch0 + 5] + beta[ch0 + 5]);
  o8[6] = f2bf((x1.z - mu) * rs * gamma[ch0 + 6] + beta[ch0 + 6]);
  o8[7] = f2bf((x1.w - mu) * rs * gamma[ch0 + 7] + beta[ch0 + 7]);
  *(v8bf*)(xnb + ((size_t)b * NPIX + n) * CH + ch0) = o8;
}

// ---------------------------------------------------------------------------
// Kernel 4: QKV GEMM, fully direct fragments (A = xnb bf16, B = wT bf16).
// 128x128 tiles, 8 waves (4 M x 2 N).  q stored pre-scaled by 1/sqrt(C);
// v stored TRANSPOSED ([C][N]); k stored row-major ([N][C]).
// grid = (NPIX/128, CH/128, BATCH*3)
// ---------------------------------------------------------------------------
__global__ void k_gemm_qkv(const bf16* __restrict__ xnb,
                           const bf16* __restrict__ wqT, const bf16* __restrict__ wkT,
                           const bf16* __restrict__ wvT,
                           const float* __restrict__ bq, const float* __restrict__ bk,
                           const float* __restrict__ bv,
                           bf16* __restrict__ qo, bf16* __restrict__ ko,
                           bf16* __restrict__ vTo) {
  const int b     = blockIdx.z / 3;
  const int which = blockIdx.z % 3;
  const int row0  = blockIdx.x * 128;
  const int col0  = blockIdx.y * 128;
  const int tid   = threadIdx.x;
  const int wid   = tid >> 5;
  const int lane  = tid & 31, hl = lane >> 4, nl = lane & 15;
  const int wm    = wid & 3, wn = wid >> 2;

  const bf16*  wsel = (which == 0) ? wqT : (which == 1) ? wkT : wvT;
  const float* bsel = (which == 0) ? bq : (which == 1) ? bk : bv;
  const bf16*  ab   = xnb + (size_t)b * NPIX * CH;

  v8f acc[2][4];
  const v8f vz = {};
#pragma unroll
  for (int i = 0; i < 2; ++i)
#pragma unroll
    for (int j = 0; j < 4; ++j) acc[i][j] = vz;

  for (int kk = 0; kk < CH; kk += 32) {
    const v16bf a0 = load_a_frag(ab, CH, row0 + wm * 32, kk);
    const v16bf a1 = load_a_frag(ab, CH, row0 + wm * 32 + 16, kk);
#pragma unroll
    for (int j = 0; j < 4; ++j) {
      const v16bf bb = load_b_frag_nmaj(wsel, CH, kk, col0 + wn * 64 + j * 16);
      acc[0][j] = wmma_bf16(a0, bb, acc[0][j]);
      acc[1][j] = wmma_bf16(a1, bb, acc[1][j]);
    }
  }

#pragma unroll
  for (int i = 0; i < 2; ++i) {
#pragma unroll
    for (int j = 0; j < 4; ++j) {
      const int col = col0 + wn * 64 + j * 16 + nl;
      const float bias = bsel[col];
      if (which == 2) {
        // vT[ch][key]: consecutive accumulator rows = consecutive keys.
        v8bf pack;
#pragma unroll
        for (int r = 0; r < 8; ++r) pack[r] = f2bf(acc[i][j][r] + bias);
        const int key0 = row0 + wm * 32 + i * 16 + 8 * hl;
        *(v8bf*)(vTo + ((size_t)b * CH + col) * NPIX + key0) = pack;
      } else {
#pragma unroll
        for (int r = 0; r < 8; ++r) {
          const int row = row0 + wm * 32 + i * 16 + r + 8 * hl;
          const float val = acc[i][j][r] + bias;
          if (which == 0)
            qo[((size_t)b * NPIX + row) * CH + col] = f2bf(val * QSCALE);
          else
            ko[((size_t)b * NPIX + row) * CH + col] = f2bf(val);
        }
      }
    }
  }
}

// ---------------------------------------------------------------------------
// Kernel 5: flash attention. Block = 32 queries, 8 waves, 64-key tiles.
//  - Q tile (32x512 bf16 = 32 KB) staged ONCE into LDS via async-to-LDS and
//    reused for all 64 key tiles.
//  - S phase: S[32 x 64] as a 2x4 fragment grid, one 16x16 tile per wave.
//  - online softmax per row (threads 0..31).
//  - PV phase: each wave owns a 64-channel slice of O (8 f32 accum frags),
//    rescales by alpha and accumulates P(LDS) @ V (vT, contiguous frags).
// grid = (NPIX/32, BATCH)
// ---------------------------------------------------------------------------
__global__ void k_attn(const bf16* __restrict__ q, const bf16* __restrict__ k,
                       const bf16* __restrict__ vT, bf16* __restrict__ ao) {
  const int b    = blockIdx.y;
  const int q0   = blockIdx.x * 32;
  const int tid  = threadIdx.x;
  const int wid  = tid >> 5;
  const int lane = tid & 31, hl = lane >> 4, nl = lane & 15;
  const int sm   = wid & 1, sn = wid >> 1;   // S-tile assignment (2 x 4)
  const int cw   = wid * 64;                 // this wave's output channel slice

  __shared__ __align__(16) bf16 sQ[32 * CH];   // 32 KB
  __shared__ float sS[32 * 68];
  __shared__ __align__(16) bf16 sP[32 * 72];
  __shared__ float sM[32], sL[32], sAl[32];

  const bf16* qb  = q  + (size_t)b * NPIX * CH;
  const bf16* kb  = k  + (size_t)b * NPIX * CH;
  const bf16* vTb = vT + (size_t)b * CH * NPIX;

  // ---- stage Q tile once (2048 x 16B segments) ----
#if HAVE_ASYNC_LDS
  for (int seg = tid; seg < 32 * 64; seg += 256) {
    const int r = seg >> 6, s8 = seg & 63;
    const bf16* src = qb + (size_t)(q0 + r) * CH + s8 * 8;
    bf16* dst = sQ + r * CH + s8 * 8;
    __builtin_amdgcn_global_load_async_to_lds_b128(
        (as1_v4i_p)(void*)src, (as3_v4i_p)(void*)dst, 0, 0);
  }
  __builtin_amdgcn_s_wait_asynccnt(0);
#else
  for (int seg = tid; seg < 32 * 64; seg += 256) {
    const int r = seg >> 6, s8 = seg & 63;
    *(v8bf*)(sQ + r * CH + s8 * 8) =
        *(const v8bf*)(qb + (size_t)(q0 + r) * CH + s8 * 8);
  }
#endif

  v8f o[2][4];
  const v8f vz = {};
#pragma unroll
  for (int i = 0; i < 2; ++i)
#pragma unroll
    for (int j = 0; j < 4; ++j) o[i][j] = vz;

  if (tid < 32) { sM[tid] = -3.0e38f; sL[tid] = 0.f; }
  __syncthreads();

  for (int kt = 0; kt < NPIX; kt += 64) {
    // ---- scores: this wave's 16x16 tile of S[32 x 64] ----
    v8f s = vz;
    for (int kk = 0; kk < CH; kk += 32) {
      const v16bf a  = load_a_frag(sQ, CH, sm * 16, kk);
      const v16bf bb = load_b_frag_nmaj(kb, CH, kk, kt + sn * 16);
      s = wmma_bf16(a, bb, s);
    }
#pragma unroll
    for (int r = 0; r < 8; ++r)
      sS[(sm * 16 + r + 8 * hl) * 68 + sn * 16 + nl] = s[r];
    __syncthreads();

    // ---- online softmax update per row ----
    if (tid < 32) {
      const float mold = sM[tid];
      float mnew = mold;
      for (int j = 0; j < 64; ++j) mnew = fmaxf(mnew, sS[tid * 68 + j]);
      const float alpha = __expf(mold - mnew);
      float lsum = 0.f;
      for (int j = 0; j < 64; ++j) {
        const float p = __expf(sS[tid * 68 + j] - mnew);
        lsum += p;
        sP[tid * 72 + j] = f2bf(p);
      }
      sM[tid]  = mnew;
      sL[tid]  = sL[tid] * alpha + lsum;
      sAl[tid] = alpha;
    }
    __syncthreads();

    // ---- rescale O, accumulate P @ V over this 64-key tile ----
#pragma unroll
    for (int mf = 0; mf < 2; ++mf) {
#pragma unroll
      for (int r = 0; r < 8; ++r) {
        const float a = sAl[mf * 16 + r + 8 * hl];
#pragma unroll
        for (int nf = 0; nf < 4; ++nf) o[mf][nf][r] *= a;
      }
      const v16bf pa0 = load_a_frag(sP, 72, mf * 16, 0);
      const v16bf pa1 = load_a_frag(sP, 72, mf * 16, 32);
#pragma unroll
      for (int nf = 0; nf < 4; ++nf) {
        const v16bf vf0 = load_b_frag_nmaj(vTb, NPIX, kt,      cw + nf * 16);
        const v16bf vf1 = load_b_frag_nmaj(vTb, NPIX, kt + 32, cw + nf * 16);
        o[mf][nf] = wmma_bf16(pa0, vf0, o[mf][nf]);
        o[mf][nf] = wmma_bf16(pa1, vf1, o[mf][nf]);
      }
    }
    __syncthreads();
  }

  // ---- finalize: divide by softmax denominator, store bf16 ----
  bf16* aob = ao + (size_t)b * NPIX * CH;
#pragma unroll
  for (int mf = 0; mf < 2; ++mf) {
#pragma unroll
    for (int r = 0; r < 8; ++r) {
      const int row = mf * 16 + r + 8 * hl;
      const float inv = 1.0f / sL[row];
#pragma unroll
      for (int nf = 0; nf < 4; ++nf)
        aob[(size_t)(q0 + row) * CH + cw + nf * 16 + nl] = f2bf(o[mf][nf][r] * inv);
    }
  }
}

// ---------------------------------------------------------------------------
// Kernel 6: projection GEMM + bias + residual (f32 out).
// Direct-from-global fragments (A = attention output bf16, B = WpT bf16).
// grid = (NPIX/128, CH/128, BATCH)
// ---------------------------------------------------------------------------
__global__ void k_proj(const bf16* __restrict__ ao, const bf16* __restrict__ wpT,
                       const float* __restrict__ bp, const float* __restrict__ x,
                       float* __restrict__ out) {
  const int b    = blockIdx.z;
  const int row0 = blockIdx.x * 128;
  const int col0 = blockIdx.y * 128;
  const int tid  = threadIdx.x;
  const int wid  = tid >> 5;
  const int lane = tid & 31, hl = lane >> 4, nl = lane & 15;
  const int wm   = wid & 3, wn = wid >> 2;

  const bf16* aob = ao + (size_t)b * NPIX * CH;

  v8f acc[2][4];
  const v8f vz = {};
#pragma unroll
  for (int i = 0; i < 2; ++i)
#pragma unroll
    for (int j = 0; j < 4; ++j) acc[i][j] = vz;

  for (int kk = 0; kk < CH; kk += 32) {
    __builtin_prefetch(aob + (size_t)(row0 + wm * 32) * CH + kk + 64, 0, 1);
    const v16bf a0 = load_a_frag(aob, CH, row0 + wm * 32, kk);
    const v16bf a1 = load_a_frag(aob, CH, row0 + wm * 32 + 16, kk);
#pragma unroll
    for (int j = 0; j < 4; ++j) {
      const v16bf bb = load_b_frag_nmaj(wpT, CH, kk, col0 + wn * 64 + j * 16);
      acc[0][j] = wmma_bf16(a0, bb, acc[0][j]);
      acc[1][j] = wmma_bf16(a1, bb, acc[1][j]);
    }
  }

#pragma unroll
  for (int i = 0; i < 2; ++i) {
#pragma unroll
    for (int j = 0; j < 4; ++j) {
      const int col = col0 + wn * 64 + j * 16 + nl;
      const float bias = bp[col];
#pragma unroll
      for (int r = 0; r < 8; ++r) {
        const int row = row0 + wm * 32 + i * 16 + r + 8 * hl;
        const size_t idx = ((size_t)b * NPIX + row) * CH + col;
        out[idx] = acc[i][j][r] + bias + x[idx];
      }
    }
  }
}

// ---------------------------------------------------------------------------
// Host launcher
// ---------------------------------------------------------------------------
extern "C" void kernel_launch(void* const* d_in, const int* in_sizes, int n_in,
                              void* d_out, int out_size, void* d_ws, size_t ws_size,
                              hipStream_t stream) {
  (void)in_sizes; (void)n_in; (void)out_size; (void)ws_size;

  const float* x     = (const float*)d_in[0];
  const float* gamma = (const float*)d_in[1];
  const float* beta  = (const float*)d_in[2];
  const float* Wq    = (const float*)d_in[3];
  const float* bq    = (const float*)d_in[4];
  const float* Wk    = (const float*)d_in[5];
  const float* bk    = (const float*)d_in[6];
  const float* Wv    = (const float*)d_in[7];
  const float* bv    = (const float*)d_in[8];
  const float* Wp    = (const float*)d_in[9];
  const float* bp    = (const float*)d_in[10];
  float* out = (float*)d_out;

  char* ws = (char*)d_ws;
  size_t off = 0;
  auto take = [&](size_t bytes) -> char* {
    char* p = ws + off;
    off += (bytes + 255) & ~(size_t)255;
    return p;
  };

  float* stats = (float*)take((size_t)BATCH * NG * 2 * sizeof(float));
  bf16*  wqT   = (bf16*)take((size_t)CH * CH * sizeof(bf16));
  bf16*  wkT   = (bf16*)take((size_t)CH * CH * sizeof(bf16));
  bf16*  wvT   = (bf16*)take((size_t)CH * CH * sizeof(bf16));
  bf16*  wpT   = (bf16*)take((size_t)CH * CH * sizeof(bf16));
  bf16*  xnb   = (bf16*)take((size_t)BATCH * NPIX * CH * sizeof(bf16));
  bf16*  qb    = (bf16*)take((size_t)BATCH * NPIX * CH * sizeof(bf16));
  bf16*  kb    = (bf16*)take((size_t)BATCH * NPIX * CH * sizeof(bf16));
  bf16*  vTb   = (bf16*)take((size_t)BATCH * NPIX * CH * sizeof(bf16));
  bf16*  aob   = (bf16*)take((size_t)BATCH * NPIX * CH * sizeof(bf16));

  k_convert_w<<<dim3(CH * CH / 256, 4), 256, 0, stream>>>(Wq, Wk, Wv, Wp,
                                                          wqT, wkT, wvT, wpT);
  k_gn_stats<<<BATCH * NG, 256, 0, stream>>>(x, stats);
  k_gn_apply<<<(BATCH * NPIX * 64) / 256, 256, 0, stream>>>(x, stats, gamma, beta, xnb);
  k_gemm_qkv<<<dim3(NPIX / 128, CH / 128, BATCH * 3), 256, 0, stream>>>(
      xnb, wqT, wkT, wvT, bq, bk, bv, qb, kb, vTb);
  k_attn<<<dim3(NPIX / 32, BATCH), 256, 0, stream>>>(qb, kb, vTb, aob);
  k_proj<<<dim3(NPIX / 128, CH / 128, BATCH), 256, 0, stream>>>(aob, wpT, bp, x, out);
}
